// MHA_81887846465697
// MI455X (gfx1250) — matrix operand order repaired and checked by reading, not compiled, not run on hardware
//
#include <hip/hip_runtime.h>
#include <hip/hip_bf16.h>
#include <cstdint>

typedef unsigned short u16_t;
typedef __attribute__((ext_vector_type(16))) __bf16 v16bf;
typedef __attribute__((ext_vector_type(8)))  float  v8f;

union BF16x16 { v16bf v; uint4 q[2]; };

__device__ __forceinline__ u16_t f2bf(float f) {
    __bf16 h = (__bf16)f;                       // native RTNE cvt on gfx1250
    return __builtin_bit_cast(u16_t, h);
}
__device__ __forceinline__ float bf2f(u16_t s) {
    union { unsigned u; float f; } x; x.u = ((unsigned)s) << 16;
    return x.f;
}

#define S_LEN 2048
#define DMODEL 1024
#define NHEAD 16
#define DK 64
#define MTOT 8192          // B*S
// GEMM tiling
#define KCG 64             // staged K-chunk for the projection GEMMs (2 WMMA k-steps)
#define LDA 72             // padded LDS row stride (u16) for 64-wide rows
// attention tiling
#define KC 32              // key chunk (== WMMA K)
#define LDP 40             // padded LDS row stride (u16) for 32-wide rows
#define LDK 72

// ---------------------------------------------------------------------------
// Kernel 1: QKV projections.  C = A(fp32 MxK) * W(fp32 KxN) -> bf16 stored
// permuted as [B][H][S][DK].  blockIdx.z selects q/k/v.
// ---------------------------------------------------------------------------
__global__ __launch_bounds__(256)
void mha_qkv_gemm(const float* __restrict__ Aq, const float* __restrict__ Ak,
                  const float* __restrict__ Av,
                  const float* __restrict__ Wq, const float* __restrict__ Wk,
                  const float* __restrict__ Wv,
                  u16_t* __restrict__ Qh, u16_t* __restrict__ Kh,
                  u16_t* __restrict__ Vh)
{
    __shared__ u16_t As[128 * LDA];
    __shared__ u16_t Bs[128 * LDA];

    const int z = blockIdx.z;
    const float* A = (z == 0) ? Aq : (z == 1) ? Ak : Av;
    const float* W = (z == 0) ? Wq : (z == 1) ? Wk : Wv;
    u16_t*       O = (z == 0) ? Qh : (z == 1) ? Kh : Vh;

    const int tid   = threadIdx.x;
    const int lane  = tid & 31;
    const int wv    = tid >> 5;
    const int waveM = (wv & 3) * 32;
    const int waveN = (wv >> 2) * 64;
    const int mBase = blockIdx.y * 128;
    const int nBase = blockIdx.x * 128;
    const int m2    = lane & 15;
    const int hf    = (lane >> 4) & 1;

    v8f c[2][4] = {};

    for (int kt = 0; kt < DMODEL / KCG; ++kt) {
        // stage A tile 128x64 fp32 -> bf16 (16 float4 per row)
#pragma unroll
        for (int i = 0; i < 8; ++i) {
            int idx = tid + i * 256;
            int row = idx >> 4, j = idx & 15;
            float4 f = *(const float4*)&A[(size_t)(mBase + row) * DMODEL + kt * KCG + j * 4];
            unsigned lo = (unsigned)f2bf(f.x) | ((unsigned)f2bf(f.y) << 16);
            unsigned hi = (unsigned)f2bf(f.z) | ((unsigned)f2bf(f.w) << 16);
            *(uint2*)&As[row * LDA + j * 4] = make_uint2(lo, hi);
        }
        // stage W tile 64x128 fp32 -> bf16, transposed (n-major)
#pragma unroll
        for (int i = 0; i < 8; ++i) {
            int idx = tid + i * 256;
            int kr = idx >> 5, j = idx & 31;
            float4 f = *(const float4*)&W[(size_t)(kt * KCG + kr) * DMODEL + nBase + j * 4];
            Bs[(j * 4 + 0) * LDA + kr] = f2bf(f.x);
            Bs[(j * 4 + 1) * LDA + kr] = f2bf(f.y);
            Bs[(j * 4 + 2) * LDA + kr] = f2bf(f.z);
            Bs[(j * 4 + 3) * LDA + kr] = f2bf(f.w);
        }
        __syncthreads();

#pragma unroll
        for (int kk = 0; kk < 2; ++kk) {
            BF16x16 a[2], b[4];
#pragma unroll
            for (int mi = 0; mi < 2; ++mi) {
                const u16_t* p = &As[(waveM + mi * 16 + m2) * LDA + kk * 32 + hf * 8];
                a[mi].q[0] = *(const uint4*)p;
                a[mi].q[1] = *(const uint4*)(p + 16);
            }
#pragma unroll
            for (int ni = 0; ni < 4; ++ni) {
                const u16_t* p = &Bs[(waveN + ni * 16 + m2) * LDA + kk * 32 + hf * 8];
                b[ni].q[0] = *(const uint4*)p;
                b[ni].q[1] = *(const uint4*)(p + 16);
            }
#pragma unroll
            for (int mi = 0; mi < 2; ++mi)
#pragma unroll
                for (int ni = 0; ni < 4; ++ni)
                    c[mi][ni] = __builtin_amdgcn_wmma_f32_16x16x32_bf16(
                        false, a[mi].v, false, b[ni].v, (short)0, c[mi][ni], false, false);
        }
        __syncthreads();
    }

    // store bf16, permuted [B][H][S][DK]
#pragma unroll
    for (int mi = 0; mi < 2; ++mi) {
        int mrow = mBase + waveM + mi * 16 + hf * 8;
#pragma unroll
        for (int ni = 0; ni < 4; ++ni) {
            int n = nBase + waveN + ni * 16 + m2;
            int h = n >> 6, d = n & 63;
#pragma unroll
            for (int r = 0; r < 8; ++r) {
                int mm = mrow + r;
                int bb = mm >> 11, ss = mm & (S_LEN - 1);
                O[(((size_t)bb * NHEAD + h) * S_LEN + ss) * DK + d] = f2bf(c[mi][ni][r]);
            }
        }
    }
}

// ---------------------------------------------------------------------------
// Kernel 2: RoPE in place on Qh, Kh ([B][H][S][64] bf16).
// ---------------------------------------------------------------------------
__global__ __launch_bounds__(256)
void mha_rope(u16_t* __restrict__ Qh, u16_t* __restrict__ Kh)
{
    unsigned idx = blockIdx.x * 256u + threadIdx.x;   // 2^23 threads total
    int i  = idx & 31;
    int s  = (idx >> 5) & (S_LEN - 1);
    int bh = (idx >> 16) & 63;
    int t  = idx >> 22;
    u16_t* X = t ? Kh : Qh;
    size_t off = ((size_t)bh * S_LEN + s) * DK + 2 * i;
    // inv_freq = 10000^(-2i/64)  => exp2(-i * log2(10000)/32)
    float invf = exp2f(-(float)i * (13.287712379549449f / 32.0f));
    float ang  = (float)s * invf;
    float sn, cs;
    __sincosf(ang, &sn, &cs);
    float x1 = bf2f(X[off]), x2 = bf2f(X[off + 1]);
    X[off]     = f2bf(x1 * cs - x2 * sn);
    X[off + 1] = f2bf(x1 * sn + x2 * cs);
}

// ---------------------------------------------------------------------------
// Kernel 3: flash attention.  block = 8 waves, each wave owns a 16-row q-tile;
// K chunk staged via async global->LDS, V transposed into LDS per chunk.
// Output bf16 [B][S][H*64].
// ---------------------------------------------------------------------------
__global__ __launch_bounds__(256)
void mha_attn(const u16_t* __restrict__ Qh, const u16_t* __restrict__ Kh,
              const u16_t* __restrict__ Vh, u16_t* __restrict__ Oh)
{
    __shared__ u16_t Ks[32 * LDK];     // K chunk, row-major [kpos][64]
    __shared__ u16_t Vt[64 * LDP];     // V chunk transposed [d][kpos]
    __shared__ u16_t Pw[8][16 * LDP];  // per-wave P tile

    const int tid  = threadIdx.x, lane = tid & 31, wv = tid >> 5;
    const int bh   = blockIdx.y;
    const size_t base = (size_t)bh * S_LEN * DK;
    const int qRow0 = blockIdx.x * 128 + wv * 16;
    const int m2 = lane & 15, hf = (lane >> 4) & 1;

    // resident Q fragments (A layout), d split 0..31 / 32..63
    BF16x16 aQ[2];
#pragma unroll
    for (int di = 0; di < 2; ++di) {
        const u16_t* p = &Qh[base + (size_t)(qRow0 + m2) * DK + di * 32 + hf * 8];
        aQ[di].q[0] = *(const uint4*)p;
        aQ[di].q[1] = *(const uint4*)(p + 16);
    }

    v8f Ofr[4] = {};
    float m8[8], l8[8];
#pragma unroll
    for (int r = 0; r < 8; ++r) { m8[r] = -1e30f; l8[r] = 0.0f; }

    for (int ck = 0; ck < S_LEN / KC; ++ck) {
        const int kp0 = ck * KC;
        __syncthreads();
        {   // async stage of K chunk (32x64 bf16): each thread moves 16B
            int kl = tid >> 3, g = tid & 7;
            const u16_t* gp = &Kh[base + (size_t)(kp0 + kl) * DK + g * 8];
            unsigned lds = (unsigned)(uintptr_t)&Ks[kl * LDK + g * 8];
            asm volatile("global_load_async_to_lds_b128 %0, %1, off"
                         :: "v"(lds), "v"((unsigned long long)(uintptr_t)gp)
                         : "memory");
        }
        {   // transpose-stage V chunk: Vt[d][kpos]
            int kl = tid & 31, dg = tid >> 5;
            uint4 vv = *(const uint4*)&Vh[base + (size_t)(kp0 + kl) * DK + dg * 8];
            const u16_t* us = (const u16_t*)&vv;
#pragma unroll
            for (int j = 0; j < 8; ++j) Vt[(dg * 8 + j) * LDP + kl] = us[j];
        }
        asm volatile("s_wait_asynccnt 0" ::: "memory");
        __syncthreads();

        // scores S = Q * K^T   (16 x 32 tile, fp32 accum)
        v8f sc[2] = {};
#pragma unroll
        for (int nj = 0; nj < 2; ++nj)
#pragma unroll
            for (int di = 0; di < 2; ++di) {
                BF16x16 bk;
                const u16_t* p = &Ks[(nj * 16 + m2) * LDK + di * 32 + hf * 8];
                bk.q[0] = *(const uint4*)p;
                bk.q[1] = *(const uint4*)(p + 16);
                sc[nj] = __builtin_amdgcn_wmma_f32_16x16x32_bf16(
                    false, aQ[di].v, false, bk.v, (short)0, sc[nj], false, false);
            }

        // online softmax (row stats are uniform across each 16-lane half)
        float lm[8], fac[8], ls[8];
#pragma unroll
        for (int r = 0; r < 8; ++r) {
            sc[0][r] *= 0.125f;  sc[1][r] *= 0.125f;   // 1/sqrt(64)
            lm[r] = fmaxf(sc[0][r], sc[1][r]);
        }
#pragma unroll
        for (int off = 1; off < 16; off <<= 1)
#pragma unroll
            for (int r = 0; r < 8; ++r)
                lm[r] = fmaxf(lm[r], __shfl_xor(lm[r], off, 32));
#pragma unroll
        for (int r = 0; r < 8; ++r) {
            float nm = fmaxf(m8[r], lm[r]);
            fac[r] = __expf(m8[r] - nm);
            m8[r]  = nm;
            float p0 = __expf(sc[0][r] - nm);
            float p1 = __expf(sc[1][r] - nm);
            sc[0][r] = p0; sc[1][r] = p1;
            ls[r] = p0 + p1;
        }
#pragma unroll
        for (int off = 1; off < 16; off <<= 1)
#pragma unroll
            for (int r = 0; r < 8; ++r)
                ls[r] += __shfl_xor(ls[r], off, 32);
#pragma unroll
        for (int r = 0; r < 8; ++r) l8[r] = l8[r] * fac[r] + ls[r];
#pragma unroll
        for (int ni = 0; ni < 4; ++ni)
#pragma unroll
            for (int r = 0; r < 8; ++r) Ofr[ni][r] *= fac[r];

        // P (C-layout) -> LDS -> A-fragment  (wave-private, DS in-order)
        u16_t* P = Pw[wv];
#pragma unroll
        for (int nj = 0; nj < 2; ++nj)
#pragma unroll
            for (int r = 0; r < 8; ++r)
                P[(r + hf * 8) * LDP + nj * 16 + m2] = f2bf(sc[nj][r]);
        BF16x16 aP;
        {
            const u16_t* p = &P[m2 * LDP + hf * 8];
            aP.q[0] = *(const uint4*)p;
            aP.q[1] = *(const uint4*)(p + 16);
        }
        // O += P * V
#pragma unroll
        for (int ni = 0; ni < 4; ++ni) {
            BF16x16 bv;
            const u16_t* p = &Vt[(ni * 16 + m2) * LDP + hf * 8];
            bv.q[0] = *(const uint4*)p;
            bv.q[1] = *(const uint4*)(p + 16);
            Ofr[ni] = __builtin_amdgcn_wmma_f32_16x16x32_bf16(
                false, aP.v, false, bv.v, (short)0, Ofr[ni], false, false);
        }
    }

    // normalize + store [B][S][H*64] bf16
    const int b = bh >> 4, h = bh & 15;
#pragma unroll
    for (int r = 0; r < 8; ++r) {
        float inv = 1.0f / l8[r];
        int s = qRow0 + r + hf * 8;
#pragma unroll
        for (int ni = 0; ni < 4; ++ni) {
            int col = h * DK + ni * 16 + m2;
            Oh[((size_t)b * S_LEN + s) * DMODEL + col] = f2bf(Ofr[ni][r] * inv);
        }
    }
}

// ---------------------------------------------------------------------------
// Kernel 4: output projection.  bf16 A [8192x1024] * fp32 Wo -> fp32 out.
// A tile staged with async global->LDS copies (no conversion needed).
// ---------------------------------------------------------------------------
__global__ __launch_bounds__(256)
void mha_out_gemm(const u16_t* __restrict__ A, const float* __restrict__ W,
                  float* __restrict__ Out)
{
    __shared__ u16_t As[128 * LDA];
    __shared__ u16_t Bs[128 * LDA];

    const int tid   = threadIdx.x;
    const int lane  = tid & 31;
    const int wv    = tid >> 5;
    const int waveM = (wv & 3) * 32;
    const int waveN = (wv >> 2) * 64;
    const int mBase = blockIdx.y * 128;
    const int nBase = blockIdx.x * 128;
    const int m2    = lane & 15;
    const int hf    = (lane >> 4) & 1;

    v8f c[2][4] = {};

    for (int kt = 0; kt < DMODEL / KCG; ++kt) {
        // stage A tile 128x64 bf16 via async copies: 4 x 16B per thread
#pragma unroll
        for (int i = 0; i < 4; ++i) {
            int idx = tid + i * 256;
            int row = idx >> 3, j = idx & 7;
            const u16_t* gp = &A[(size_t)(mBase + row) * DMODEL + kt * KCG + j * 8];
            unsigned lds = (unsigned)(uintptr_t)&As[row * LDA + j * 8];
            asm volatile("global_load_async_to_lds_b128 %0, %1, off"
                         :: "v"(lds), "v"((unsigned long long)(uintptr_t)gp)
                         : "memory");
        }
        // stage W tile 64x128 fp32 -> bf16, transposed (n-major)
#pragma unroll
        for (int i = 0; i < 8; ++i) {
            int idx = tid + i * 256;
            int kr = idx >> 5, j = idx & 31;
            float4 f = *(const float4*)&W[(size_t)(kt * KCG + kr) * DMODEL + nBase + j * 4];
            Bs[(j * 4 + 0) * LDA + kr] = f2bf(f.x);
            Bs[(j * 4 + 1) * LDA + kr] = f2bf(f.y);
            Bs[(j * 4 + 2) * LDA + kr] = f2bf(f.z);
            Bs[(j * 4 + 3) * LDA + kr] = f2bf(f.w);
        }
        asm volatile("s_wait_asynccnt 0" ::: "memory");
        __syncthreads();

#pragma unroll
        for (int kk = 0; kk < 2; ++kk) {
            BF16x16 a[2], b[4];
#pragma unroll
            for (int mi = 0; mi < 2; ++mi) {
                const u16_t* p = &As[(waveM + mi * 16 + m2) * LDA + kk * 32 + hf * 8];
                a[mi].q[0] = *(const uint4*)p;
                a[mi].q[1] = *(const uint4*)(p + 16);
            }
#pragma unroll
            for (int ni = 0; ni < 4; ++ni) {
                const u16_t* p = &Bs[(waveN + ni * 16 + m2) * LDA + kk * 32 + hf * 8];
                b[ni].q[0] = *(const uint4*)p;
                b[ni].q[1] = *(const uint4*)(p + 16);
            }
#pragma unroll
            for (int mi = 0; mi < 2; ++mi)
#pragma unroll
                for (int ni = 0; ni < 4; ++ni)
                    c[mi][ni] = __builtin_amdgcn_wmma_f32_16x16x32_bf16(
                        false, a[mi].v, false, b[ni].v, (short)0, c[mi][ni], false, false);
        }
        __syncthreads();
    }

#pragma unroll
    for (int mi = 0; mi < 2; ++mi) {
        int mrow = mBase + waveM + mi * 16 + hf * 8;
#pragma unroll
        for (int ni = 0; ni < 4; ++ni) {
            int n = nBase + waveN + ni * 16 + m2;
#pragma unroll
            for (int r = 0; r < 8; ++r)
                Out[(size_t)(mrow + r) * DMODEL + n] = c[mi][ni][r];
        }
    }
}

// ---------------------------------------------------------------------------
extern "C" void kernel_launch(void* const* d_in, const int* in_sizes, int n_in,
                              void* d_out, int out_size, void* d_ws, size_t ws_size,
                              hipStream_t stream)
{
    const float* q  = (const float*)d_in[0];
    const float* k  = (const float*)d_in[1];
    const float* v  = (const float*)d_in[2];
    const float* Wq = (const float*)d_in[3];
    const float* Wk = (const float*)d_in[4];
    const float* Wv = (const float*)d_in[5];
    const float* Wo = (const float*)d_in[6];

    u16_t* ws = (u16_t*)d_ws;
    const size_t T = (size_t)4 * NHEAD * S_LEN * DK;   // 8,388,608 elements
    u16_t* Qh = ws;
    u16_t* Kh = ws + T;
    u16_t* Vh = ws + 2 * T;
    u16_t* AO = ws + 3 * T;                             // attn out [B][S][1024] bf16

    mha_qkv_gemm<<<dim3(DMODEL / 128, MTOT / 128, 3), 256, 0, stream>>>(
        q, k, v, Wq, Wk, Wv, Qh, Kh, Vh);
    mha_rope<<<(2u * 64u * S_LEN * 32u) / 256u, 256, 0, stream>>>(Qh, Kh);
    mha_attn<<<dim3(S_LEN / 128, 4 * NHEAD), 256, 0, stream>>>(Qh, Kh, Vh, AO);
    mha_out_gemm<<<dim3(DMODEL / 128, MTOT / 128), 256, 0, stream>>>(AO, Wo, (float*)d_out);
}
